// PathFormer_allin1_88957362635006
// MI455X (gfx1250) — compile-verified, matrix operand order checked by reading, not run on hardware
//
#include <hip/hip_runtime.h>

// ---------------------------------------------------------------------------
// Performer (FAVOR+) self-attention block for MI455X (gfx1250, wave32, WMMA)
// B=4, N=8192, D=512, H=8, DH=64, R=266 (padded to 288), MLP=1024
// All GEMM operands are f16 with contiguous-K layout -> pure async-DMA staging.
// ---------------------------------------------------------------------------

typedef __attribute__((ext_vector_type(16))) _Float16 v16h;
typedef __attribute__((ext_vector_type(8)))  _Float16 v8h;
typedef __attribute__((ext_vector_type(4)))  _Float16 v4h;
typedef __attribute__((ext_vector_type(4)))  float    v4f;
typedef __attribute__((ext_vector_type(8)))  float    v8f;

union F16Frag { v16h v; v8h h[2]; };
union AccFrag { v8f  v; float f[8]; };

#define R_FEAT 266
#define R_PAD  288           // multiple of 32 -> aligned K/M for WMMA tiling

__device__ __forceinline__ void stv(float* p, float v)    { *p = v; }
__device__ __forceinline__ void stv(_Float16* p, float v) { *p = (_Float16)v; }

__device__ __forceinline__ v8h zero8() {
    v8h z;
    #pragma unroll
    for (int j = 0; j < 8; ++j) z[j] = (_Float16)0.f;
    return z;
}

__device__ __forceinline__ float gelu_exact(float x) {
    return 0.5f * x * (1.0f + erff(x * 0.70710678118654752f));
}

// --- CDNA5 async global->LDS copy (ASYNCcnt path, CDNA5 ISA) ----------
__device__ __forceinline__ void async_copy_b128(const void* gptr, void* lptr) {
    unsigned           lds = (unsigned)(uintptr_t)lptr;           // LDS byte addr
    unsigned long long ga  = (unsigned long long)(uintptr_t)gptr; // 64-bit vaddr
    asm volatile("global_load_async_to_lds_b128 %0, %1, off"
                 :: "v"(lds), "v"(ga) : "memory");
}
__device__ __forceinline__ void async_wait0() {
    asm volatile("s_wait_asynccnt 0x0" ::: "memory");
}

// Epilogue modes
#define EPI_NONE      0
#define EPI_BIAS      1
#define EPI_BIAS_GELU 2
#define EPI_BIAS_RES  3   // + bias[col] + resid_f32[row*ldc+col]
#define EPI_ROWSCALE  4   // * extra[extraStride*z + row]

// ---------------------------------------------------------------------------
// Batched WMMA GEMM, fully async-DMA staged, double-buffered LDS.
//   C(MxN) = epi( A(MxK) @ B(KxN) ),  with B given PRE-TRANSPOSED as
//   Bt(NxK) f16 row-major (ldbT = row length of Bt).
// K must be a multiple of 32; lda/ldbT rows must be 16B aligned.
// AM/AN: compile-time "dimension covers whole 128 tile" flags; out-of-range
// LDS rows are zero-filled ONCE in the prologue (tile-invariant).
// Per-batch offsets: off = (z/IN)*OS + (z%IN)*IS (element units).
// ---------------------------------------------------------------------------
template<typename TC, int EPI, bool AM, bool AN>
__global__ __launch_bounds__(256)
void wmma_gemm_k(const _Float16* __restrict__ A, const _Float16* __restrict__ Bt,
                 TC* __restrict__ C,
                 const float* __restrict__ bias, const float* __restrict__ extra,
                 int M, int N, int K, int lda, int ldbT, int ldc,
                 long aOS, long aIS, int aIN,
                 long bOS, long bIS, int bIN,
                 long cOS, long cIS, int cIN,
                 long extraStride)
{
    __shared__ _Float16 As[2][128][40];   // [buf][m][k], +8 halves pad
    __shared__ _Float16 Bs[2][128][40];   // [buf][n][k]

    const int tid  = threadIdx.x;
    const int lane = tid & 31;
    const int wid  = tid >> 5;
    const int waveRow = (wid >> 1) * 32;
    const int waveCol = (wid & 1) * 64;
    const int bz    = blockIdx.z;
    const int mBase = blockIdx.y * 128;
    const int nBase = blockIdx.x * 128;

    const long aOff = (long)(bz / aIN) * aOS + (long)(bz % aIN) * aIS;
    const long bOff = (long)(bz / bIN) * bOS + (long)(bz % bIN) * bIS;
    const long cOff = (long)(bz / cIN) * cOS + (long)(bz % cIN) * cIS;

    AccFrag acc[2][4];
    #pragma unroll
    for (int a = 0; a < 2; ++a)
        #pragma unroll
        for (int b = 0; b < 4; ++b)
            #pragma unroll
            for (int j = 0; j < 8; ++j) acc[a][b].f[j] = 0.f;

    const int halfSel = (lane >> 4) << 3;
    const int ktiles = K >> 5;

    // stage one 128x32 tile via async DMA (valid rows only)
    auto stageA = [&](int kt, int buf) {
        const int k0 = kt << 5;
        #pragma unroll
        for (int i = 0; i < 2; ++i) {
            int id = i * 256 + tid;
            int row = id >> 2, c8 = (id & 3) * 8;
            int gm = mBase + row;
            if (AM || gm < M)
                async_copy_b128(A + aOff + (long)gm * lda + (k0 + c8), &As[buf][row][c8]);
        }
    };
    auto stageB = [&](int kt, int buf) {
        const int k0 = kt << 5;
        #pragma unroll
        for (int i = 0; i < 2; ++i) {
            int id = i * 256 + tid;
            int row = id >> 2, c8 = (id & 3) * 8;
            int gn = nBase + row;
            if (AN || gn < N)
                async_copy_b128(Bt + bOff + (long)gn * ldbT + (k0 + c8), &Bs[buf][row][c8]);
        }
    };

    // zero-fill tile-invariant out-of-range rows once (both buffers)
    if constexpr (!AM) {
        #pragma unroll
        for (int i = 0; i < 2; ++i) {
            int id = i * 256 + tid;
            int row = id >> 2, c8 = (id & 3) * 8;
            if (mBase + row >= M) {
                *(v8h*)&As[0][row][c8] = zero8();
                *(v8h*)&As[1][row][c8] = zero8();
            }
        }
    }
    if constexpr (!AN) {
        #pragma unroll
        for (int i = 0; i < 2; ++i) {
            int id = i * 256 + tid;
            int row = id >> 2, c8 = (id & 3) * 8;
            if (nBase + row >= N) {
                *(v8h*)&Bs[0][row][c8] = zero8();
                *(v8h*)&Bs[1][row][c8] = zero8();
            }
        }
    }

    // prologue: fill buffer 0
    stageA(0, 0);
    stageB(0, 0);
    async_wait0();
    __syncthreads();

    for (int kt = 0; kt < ktiles; ++kt) {
        const int cur = kt & 1;
        // prefetch next tile into the other buffer (DMA overlaps wmma below)
        if (kt + 1 < ktiles) {
            stageA(kt + 1, cur ^ 1);
            stageB(kt + 1, cur ^ 1);
        }

        F16Frag af[2], bf[4];
        #pragma unroll
        for (int sm = 0; sm < 2; ++sm) {
            int m = waveRow + sm * 16 + (lane & 15);
            af[sm].h[0] = *(const v8h*)&As[cur][m][halfSel];
            af[sm].h[1] = *(const v8h*)&As[cur][m][16 + halfSel];
        }
        #pragma unroll
        for (int sn = 0; sn < 4; ++sn) {
            int n = waveCol + sn * 16 + (lane & 15);
            bf[sn].h[0] = *(const v8h*)&Bs[cur][n][halfSel];
            bf[sn].h[1] = *(const v8h*)&Bs[cur][n][16 + halfSel];
        }
        #pragma unroll
        for (int sm = 0; sm < 2; ++sm)
            #pragma unroll
            for (int sn = 0; sn < 4; ++sn)
                acc[sm][sn].v = __builtin_amdgcn_wmma_f32_16x16x32_f16(
                    false, af[sm].v, false, bf[sn].v,
                    (short)0, acc[sm][sn].v, false, false);

        async_wait0();    // tail of next-tile DMA
        __syncthreads();
    }

    const int halfId = lane >> 4;
    const int col0   = lane & 15;
    #pragma unroll
    for (int sm = 0; sm < 2; ++sm) {
        #pragma unroll
        for (int sn = 0; sn < 4; ++sn) {
            #pragma unroll
            for (int i = 0; i < 8; ++i) {
                int gm = mBase + waveRow + sm * 16 + i + 8 * halfId;
                int gn = nBase + waveCol + sn * 16 + col0;
                if ((AM && AN) || (gm < M && gn < N)) {
                    float v = acc[sm][sn].f[i];
                    if constexpr (EPI == EPI_BIAS || EPI == EPI_BIAS_GELU || EPI == EPI_BIAS_RES)
                        v += bias[gn];
                    if constexpr (EPI == EPI_BIAS_GELU) v = gelu_exact(v);
                    if constexpr (EPI == EPI_BIAS_RES)  v += extra[(long)gm * ldc + gn];
                    if constexpr (EPI == EPI_ROWSCALE)  v *= extra[extraStride * (long)bz + gm];
                    stv(&C[cOff + (long)gm * ldc + gn], v);
                }
            }
        }
    }
}

// f32 -> f16 bulk convert (vectorized b128 load / b64 store)
__global__ __launch_bounds__(256)
void cvt16_k(const float* __restrict__ in, _Float16* __restrict__ out)
{
    long i = ((long)blockIdx.x * 256 + threadIdx.x) * 4;
    v4f g = *(const v4f*)(in + i);
    v4h h;
    #pragma unroll
    for (int j = 0; j < 4; ++j) h[j] = (_Float16)g[j];
    *(v4h*)(out + i) = h;
}

// Weight transpose + convert: W (KxN f32, row-major) -> Wt (NxK f16), LDS-tiled
__global__ __launch_bounds__(256)
void wtcvt_k(const float* __restrict__ W, _Float16* __restrict__ Wt, int K, int N)
{
    __shared__ _Float16 t[64][72];
    const int tid = threadIdx.x;
    const int kBase = blockIdx.y * 64, nBase = blockIdx.x * 64;
    #pragma unroll
    for (int i = 0; i < 16; ++i) {
        int id = i * 256 + tid;
        int r = id >> 6, c = id & 63;                       // r: k-row, c: n-col
        t[c][r] = (_Float16)W[(long)(kBase + r) * N + nBase + c];
    }
    __syncthreads();
    #pragma unroll
    for (int i = 0; i < 16; ++i) {
        int id = i * 256 + tid;
        int r = id >> 6, c = id & 63;                       // r: n-row, c: k-col
        Wt[(long)(nBase + r) * K + kBase + c] = t[r][c];
    }
}

// Per-head V transpose: v16 (B*N, 512 head cols) -> vT (BH*64, 8192), LDS-tiled
__global__ __launch_bounds__(256)
void vtrans_k(const _Float16* __restrict__ v, _Float16* __restrict__ vt)
{
    __shared__ _Float16 t[64][72];
    const int tid = threadIdx.x;
    const int z = blockIdx.y, b = z >> 3, h = z & 7;
    const int nBase = blockIdx.x * 64;
    #pragma unroll
    for (int i = 0; i < 2; ++i) {
        int id = i * 256 + tid;
        int r = id >> 3, c8 = (id & 7) * 8;                 // r: n-local, c8: d chunk
        v8h g = *(const v8h*)(v + ((long)(b * 8192 + nBase + r)) * 512 + h * 64 + c8);
        #pragma unroll
        for (int j = 0; j < 8; ++j) t[c8 + j][r] = g[j];
    }
    __syncthreads();
    #pragma unroll
    for (int i = 0; i < 2; ++i) {
        int id = i * 256 + tid;
        int r = id >> 3, c8 = (id & 7) * 8;                 // r: d row, c8: n chunk
        v8h g = *(const v8h*)&t[r][c8];
        *(v8h*)(vt + ((long)(z * 64 + r)) * 8192 + nBase + c8) = g;
    }
}

// ---------------------------------------------------------------------------
// Fused Performer feature map: one wave = 16x288 dd tile in WMMA registers.
// MODE 0: queries -> per-row stabilizer, store qp (BH, N, R_PAD), pad cols = 0
// MODE 1: keys pass-1 -> block max of dd -> partials
// MODE 2: keys pass-2 -> global stabilizer, store kp TRANSPOSED (BH, R_PAD, N)
// ---------------------------------------------------------------------------
template<int MODE>
__global__ __launch_bounds__(128)
void perf_feat_k(const _Float16* __restrict__ qk, const float* __restrict__ proj,
                 _Float16* __restrict__ outp, const float* __restrict__ stab_scalar,
                 float* __restrict__ partials)
{
    __shared__ _Float16 Ps[R_PAD][72];   // proj [r][k] (native layout), pad rows = 0
    __shared__ _Float16 Aq[4][16][72];   // 16 q rows per wave
    __shared__ float    diag_s[4][16];

    const int tid = threadIdx.x, lane = tid & 31, wid = tid >> 5;
    const int z = blockIdx.z, b = z >> 3, h = z & 7;
    const int base_n = blockIdx.x * 64 + wid * 16;

    // stage proj: 288*16 float4-chunks / 128 threads = 36 per thread
    #pragma unroll
    for (int i = 0; i < 36; ++i) {
        int id = i * 128 + tid;
        int r = id >> 4, c4 = (id & 15) * 4;
        v4h hh;
        if (r < R_FEAT) {
            v4f g = *(const v4f*)(proj + r * 64 + c4);
            #pragma unroll
            for (int j = 0; j < 4; ++j) hh[j] = (_Float16)g[j];
        } else {
            #pragma unroll
            for (int j = 0; j < 4; ++j) hh[j] = (_Float16)0.f;
        }
        *(v4h*)&Ps[r][c4] = hh;
    }
    // stage 16 q rows per wave: async DMA, 4 x 16B chunks per lane
    #pragma unroll
    for (int i = 0; i < 4; ++i) {
        int id = i * 32 + lane;
        int row = id >> 3, c8 = (id & 7) * 8;
        async_copy_b128(qk + ((long)(b * 8192 + base_n + row)) * 512 + h * 64 + c8,
                        &Aq[wid][row][c8]);
    }
    async_wait0();
    __syncthreads();

    if (MODE != 1) {
        if (lane < 16) {
            float s = 0.f;
            #pragma unroll
            for (int k = 0; k < 64; ++k) { float f = (float)Aq[wid][lane][k]; s += f * f; }
            diag_s[wid][lane] = 0.0625f * s;      // 0.5 * 64^-0.5
        }
        __syncthreads();
    }

    AccFrag acc[18];
    #pragma unroll
    for (int t = 0; t < 18; ++t)
        #pragma unroll
        for (int j = 0; j < 8; ++j) acc[t].f[j] = 0.f;

    const int halfSel = (lane >> 4) << 3;
    const int m0 = lane & 15;
    #pragma unroll
    for (int kt = 0; kt < 2; ++kt) {
        const int k0 = kt * 32;
        F16Frag af;
        af.h[0] = *(const v8h*)&Aq[wid][m0][k0 + halfSel];
        af.h[1] = *(const v8h*)&Aq[wid][m0][k0 + 16 + halfSel];
        #pragma unroll
        for (int rt = 0; rt < 18; ++rt) {
            F16Frag bf;
            bf.h[0] = *(const v8h*)&Ps[rt * 16 + m0][k0 + halfSel];
            bf.h[1] = *(const v8h*)&Ps[rt * 16 + m0][k0 + 16 + halfSel];
            acc[rt].v = __builtin_amdgcn_wmma_f32_16x16x32_f16(
                false, af.v, false, bf.v, (short)0, acc[rt].v, false, false);
        }
    }

    const float NORMF = 0.35355339059327373f;   // 64^-0.25
    const float RATIO = 0.06131393394849658f;   // 266^-0.5
    const int halfId = lane >> 4, col0 = lane & 15;

    if constexpr (MODE == 1) {
        float wmax = -3.0e38f;
        #pragma unroll
        for (int rt = 0; rt < 18; ++rt) {
            int r = rt * 16 + col0;
            if (r < R_FEAT)
                #pragma unroll
                for (int i = 0; i < 8; ++i) wmax = fmaxf(wmax, acc[rt].f[i]);
        }
        for (int o = 16; o; o >>= 1) wmax = fmaxf(wmax, __shfl_xor(wmax, o));
        if (lane == 0)
            partials[((long)z * gridDim.x + blockIdx.x) * 4 + wid] = NORMF * wmax;
        return;
    } else {
        float stab_rows[8];
        if constexpr (MODE == 0) {
            #pragma unroll
            for (int i = 0; i < 8; ++i) {
                float m = -3.0e38f;
                #pragma unroll
                for (int rt = 0; rt < 18; ++rt) {
                    int r = rt * 16 + col0;
                    if (r < R_FEAT) m = fmaxf(m, acc[rt].f[i]);
                }
                #pragma unroll
                for (int o = 8; o; o >>= 1) m = fmaxf(m, __shfl_xor(m, o));
                stab_rows[i] = NORMF * m;
            }
        }
        float stab_g = 0.f;
        if constexpr (MODE == 2) stab_g = *stab_scalar;

        #pragma unroll
        for (int i = 0; i < 8; ++i) {
            const float dg = diag_s[wid][i + 8 * halfId];
            const float st = (MODE == 0) ? stab_rows[i] : stab_g;
            const int n = base_n + i + 8 * halfId;
            #pragma unroll
            for (int rt = 0; rt < 18; ++rt) {
                int r = rt * 16 + col0;
                float dd = NORMF * acc[rt].f[i];
                float val = (r < R_FEAT) ? RATIO * (__expf(dd - dg - st) + 1e-4f) : 0.f;
                if (MODE == 0)
                    outp[((long)z * 8192 + n) * R_PAD + r] = (_Float16)val;  // qp
                else
                    outp[((long)z * R_PAD + r) * 8192 + n] = (_Float16)val;  // kp^T
            }
        }
    }
}

// ---------------------------------------------------------------------------
__global__ void gmax_k(const float* __restrict__ p, int n, float* __restrict__ out)
{
    __shared__ float sm[256];
    float m = -3.0e38f;
    for (int i = threadIdx.x; i < n; i += 256) m = fmaxf(m, p[i]);
    sm[threadIdx.x] = m;
    __syncthreads();
    for (int s = 128; s > 0; s >>= 1) {
        if ((int)threadIdx.x < s) sm[threadIdx.x] = fmaxf(sm[threadIdx.x], sm[threadIdx.x + s]);
        __syncthreads();
    }
    if (threadIdx.x == 0) *out = sm[0];
}

// ksum[z*R_PAD+r] = sum_n kp_t[z][r][n]   (one wave per (z,r) row)
__global__ __launch_bounds__(256)
void ksum_k(const _Float16* __restrict__ kp, float* __restrict__ ksum)
{
    int wid = threadIdx.x >> 5, lane = threadIdx.x & 31;
    int rid = blockIdx.x * 8 + wid;            // 0 .. 32*R_PAD-1 (exact grid)
    const _Float16* row = kp + (long)rid * 8192;
    float s = 0.f;
    for (int i = lane; i < 8192; i += 32) s += (float)row[i];
    for (int o = 16; o; o >>= 1) s += __shfl_xor(s, o);
    if (!lane) ksum[rid] = s;
}

// dinv[z*8192+n] = 1 / dot(qp[z,n,:], ksum[z,:])   (one wave per row)
__global__ __launch_bounds__(256)
void dinv_k(const _Float16* __restrict__ qp, const float* __restrict__ ksum,
            float* __restrict__ dinv)
{
    int wid = threadIdx.x >> 5, lane = threadIdx.x & 31;
    long rid = (long)blockIdx.x * 8 + wid;     // 0..262143
    int z = (int)(rid >> 13);
    float s = 0.f;
    #pragma unroll
    for (int r = lane; r < R_PAD; r += 32)     // padding contributes zeros
        s += (float)qp[rid * R_PAD + r] * ksum[z * R_PAD + r];
    for (int o = 16; o; o >>= 1) s += __shfl_xor(s, o);
    if (!lane) dinv[rid] = 1.f / s;
}

// LayerNorm over rows of 512; optionally emit f16 copy for downstream GEMMs
template<bool W16>
__global__ __launch_bounds__(256)
void ln_k(const float* __restrict__ in, const float* __restrict__ g,
          const float* __restrict__ bta, float* __restrict__ out,
          _Float16* __restrict__ out16)
{
    int wid = threadIdx.x >> 5, lane = threadIdx.x & 31;
    long row = (long)blockIdx.x * 8 + wid;
    const float* rp = in + row * 512;
    float v[16]; float s = 0.f;
    #pragma unroll
    for (int j = 0; j < 16; ++j) { v[j] = rp[lane + 32 * j]; s += v[j]; }
    for (int o = 16; o; o >>= 1) s += __shfl_xor(s, o);
    float mean = s * (1.f / 512.f);
    float q = 0.f;
    #pragma unroll
    for (int j = 0; j < 16; ++j) { float d = v[j] - mean; q += d * d; }
    for (int o = 16; o; o >>= 1) q += __shfl_xor(q, o);
    float rs = rsqrtf(q * (1.f / 512.f) + 1e-5f);
    #pragma unroll
    for (int j = 0; j < 16; ++j) {
        int c = lane + 32 * j;
        float y = (v[j] - mean) * rs * g[c] + bta[c];
        out[row * 512 + c] = y;
        if (W16) out16[row * 512 + c] = (_Float16)y;
    }
}

// ---------------------------------------------------------------------------
extern "C" void kernel_launch(void* const* d_in, const int* in_sizes, int n_in,
                              void* d_out, int out_size, void* d_ws, size_t ws_size,
                              hipStream_t stream)
{
    const float* x     = (const float*)d_in[0];
    const float* Wq    = (const float*)d_in[1];
    const float* Wk    = (const float*)d_in[2];
    const float* Wv    = (const float*)d_in[3];
    const float* Wo    = (const float*)d_in[4];
    const float* bo    = (const float*)d_in[5];
    const float* proj  = (const float*)d_in[6];
    const float* ln0g  = (const float*)d_in[7];
    const float* ln0b  = (const float*)d_in[8];
    const float* W1    = (const float*)d_in[9];
    const float* b1    = (const float*)d_in[10];
    const float* W2    = (const float*)d_in[11];
    const float* b2    = (const float*)d_in[12];
    const float* ln1g  = (const float*)d_in[13];
    const float* ln1b  = (const float*)d_in[14];

    char* ws = (char*)d_ws;
    size_t off = 0;
    auto alloc = [&](size_t bytes) { size_t o = off; off = (off + bytes + 255) & ~(size_t)255; return o; };

    const size_t SZ_ACT16 = 32768UL * 512 * 2;           // 33.5 MB
    const size_t SZ_FEAT  = 32UL * 8192 * R_PAD * 2;     // 151 MB
    const size_t SZ_ACT32 = 32768UL * 512 * 4;           // 67.1 MB

    size_t o_q16  = alloc(SZ_ACT16);
    size_t o_k16  = alloc(SZ_ACT16);
    size_t o_v16  = alloc(SZ_ACT16);
    size_t o_vt   = alloc(SZ_ACT16);                     // vT (BH*64, 8192)
    size_t o_qp   = alloc(SZ_FEAT);
    size_t o_kp   = alloc(SZ_FEAT);
    size_t o_ctx  = alloc(32UL * 64 * R_PAD * 2);        // ctxT (BH, 64, 288)
    size_t o_ksum = alloc(32UL * R_PAD * 4);
    size_t o_dinv = alloc(262144UL * 4);
    size_t o_part = alloc(16384UL * 4);
    size_t o_stab = alloc(256);
    size_t o_wqt  = alloc(512UL * 512 * 2);
    size_t o_wkt  = alloc(512UL * 512 * 2);
    size_t o_wvt  = alloc(512UL * 512 * 2);
    size_t o_wot  = alloc(512UL * 512 * 2);
    size_t o_w1t  = alloc(1024UL * 512 * 2);             // W1t (1024 x 512)
    size_t o_w2t  = alloc(512UL * 1024 * 2);             // W2t (512 x 1024)
    (void)ws_size; (void)n_in; (void)in_sizes; (void)out_size;

    _Float16* q16    = (_Float16*)(ws + o_q16);
    _Float16* k16    = (_Float16*)(ws + o_k16);
    _Float16* v16    = (_Float16*)(ws + o_v16);
    _Float16* vT     = (_Float16*)(ws + o_vt);
    _Float16* qp     = (_Float16*)(ws + o_qp);
    _Float16* kp     = (_Float16*)(ws + o_kp);
    _Float16* ctxT   = (_Float16*)(ws + o_ctx);
    float*    ksum   = (float*)(ws + o_ksum);
    float*    dinv   = (float*)(ws + o_dinv);
    float*    part   = (float*)(ws + o_part);
    float*    stab   = (float*)(ws + o_stab);
    _Float16* Wqt    = (_Float16*)(ws + o_wqt);
    _Float16* Wkt    = (_Float16*)(ws + o_wkt);
    _Float16* Wvt    = (_Float16*)(ws + o_wvt);
    _Float16* Wot    = (_Float16*)(ws + o_wot);
    _Float16* W1t    = (_Float16*)(ws + o_w1t);
    _Float16* W2t    = (_Float16*)(ws + o_w2t);
    // aliases (lifetimes disjoint in launch order below)
    _Float16* x16    = (_Float16*)(ws + o_qp);               // dead before qp written
    _Float16* attn16 = (_Float16*)(ws + o_k16);              // k16 dead after feat pass 2
    float*    res0   = (float*)(ws + o_kp);                  // kp dead after ctx GEMM
    float*    hdd    = (float*)(ws + o_kp + SZ_ACT32);
    _Float16* hdd16  = (_Float16*)(ws + o_q16);              // q16 dead after feat pass 0
    _Float16* mlph16 = (_Float16*)(ws + o_qp);               // qp dead after out GEMM
    float*    res1   = (float*)(ws + o_qp + SZ_ACT32);
    float*    outf   = (float*)d_out;

    // 0. one-time conversions: x -> f16; weights -> transposed f16
    cvt16_k<<<16384, 256, 0, stream>>>(x, x16);
    wtcvt_k<<<dim3(8, 8),  256, 0, stream>>>(Wq, Wqt, 512, 512);
    wtcvt_k<<<dim3(8, 8),  256, 0, stream>>>(Wk, Wkt, 512, 512);
    wtcvt_k<<<dim3(8, 8),  256, 0, stream>>>(Wv, Wvt, 512, 512);
    wtcvt_k<<<dim3(8, 8),  256, 0, stream>>>(Wo, Wot, 512, 512);
    wtcvt_k<<<dim3(16, 8), 256, 0, stream>>>(W1, W1t, 512, 1024);
    wtcvt_k<<<dim3(8, 16), 256, 0, stream>>>(W2, W2t, 1024, 512);

    const dim3 gQKV(4, 256, 1);
    // 1-3. q/k/v = x16 @ W  (all-async, double-buffered)
    wmma_gemm_k<_Float16, EPI_NONE, true, true><<<gQKV, 256, 0, stream>>>(
        x16, Wqt, q16, nullptr, nullptr, 32768, 512, 512, 512, 512, 512,
        0L, 0L, 1, 0L, 0L, 1, 0L, 0L, 1, 0L);
    wmma_gemm_k<_Float16, EPI_NONE, true, true><<<gQKV, 256, 0, stream>>>(
        x16, Wkt, k16, nullptr, nullptr, 32768, 512, 512, 512, 512, 512,
        0L, 0L, 1, 0L, 0L, 1, 0L, 0L, 1, 0L);
    wmma_gemm_k<_Float16, EPI_NONE, true, true><<<gQKV, 256, 0, stream>>>(
        x16, Wvt, v16, nullptr, nullptr, 32768, 512, 512, 512, 512, 512,
        0L, 0L, 1, 0L, 0L, 1, 0L, 0L, 1, 0L);

    // 3b. v -> vT per head (LDS-tiled, coalesced)
    vtrans_k<<<dim3(128, 32), 256, 0, stream>>>(v16, vT);

    // 4. qp = feature map (per-row stabilizer), padded cols zeroed
    perf_feat_k<0><<<dim3(128, 1, 32), 128, 0, stream>>>(q16, proj, qp, nullptr, nullptr);
    // 5-6. global key stabilizer
    perf_feat_k<1><<<dim3(128, 1, 32), 128, 0, stream>>>(k16, proj, nullptr, nullptr, part);
    gmax_k<<<1, 256, 0, stream>>>(part, 16384, stab);
    // 7. kp^T (BH, R_PAD, N), padded rows zeroed
    perf_feat_k<2><<<dim3(128, 1, 32), 128, 0, stream>>>(k16, proj, kp, stab, nullptr);

    // 8. ksum over N  (32*288 rows, 8 per block)
    ksum_k<<<(32 * R_PAD) / 8, 256, 0, stream>>>(kp, ksum);

    // 9. ctxT(64x288) = vT(64x8192) @ [Bt = kp (288x8192)] : batched 32x
    wmma_gemm_k<_Float16, EPI_NONE, false, false><<<dim3(3, 1, 32), 256, 0, stream>>>(
        vT, kp, ctxT, nullptr, nullptr, 64, R_PAD, 8192, 8192, 8192, R_PAD,
        64L * 8192, 0L, 1,
        (long)R_PAD * 8192, 0L, 1,
        64L * R_PAD, 0L, 1, 0L);

    // 10. d_inv
    dinv_k<<<32768, 256, 0, stream>>>(qp, ksum, dinv);

    // 11. attn = (qp @ ctx) * d_inv : A=qp(8192x288), Bt=ctxT(64x288), 32 batches
    wmma_gemm_k<_Float16, EPI_ROWSCALE, true, false><<<dim3(1, 64, 32), 256, 0, stream>>>(
        qp, ctxT, attn16, nullptr, dinv, 8192, 64, R_PAD, R_PAD, R_PAD, 512,
        8192L * R_PAD, 0L, 1,
        64L * R_PAD, 0L, 1,
        8192L * 512, 64L, 8,
        8192L);

    // 12. res0 = attn @ Wo + bo + x
    wmma_gemm_k<float, EPI_BIAS_RES, true, true><<<gQKV, 256, 0, stream>>>(
        attn16, Wot, res0, bo, x, 32768, 512, 512, 512, 512, 512,
        0L, 0L, 1, 0L, 0L, 1, 0L, 0L, 1, 0L);

    // 13. hdd = LN0(res0)  (+ f16 copy)
    ln_k<true><<<4096, 256, 0, stream>>>(res0, ln0g, ln0b, hdd, hdd16);

    // 14. mlp hidden = gelu(hdd @ W1 + b1)
    wmma_gemm_k<_Float16, EPI_BIAS_GELU, true, true><<<dim3(8, 256, 1), 256, 0, stream>>>(
        hdd16, W1t, mlph16, b1, nullptr, 32768, 1024, 512, 512, 512, 1024,
        0L, 0L, 1, 0L, 0L, 1, 0L, 0L, 1, 0L);

    // 15. res1 = mlph @ W2 + b2 + hdd
    wmma_gemm_k<float, EPI_BIAS_RES, true, true><<<gQKV, 256, 0, stream>>>(
        mlph16, W2t, res1, b2, hdd, 32768, 512, 1024, 1024, 1024, 512,
        0L, 0L, 1, 0L, 0L, 1, 0L, 0L, 1, 0L);

    // 16. out = LN1(res1)
    ln_k<false><<<4096, 256, 0, stream>>>(res1, ln1g, ln1b, outf, nullptr);
}
